// BertSelfAttention_60593398611978
// MI455X (gfx1250) — matrix operand order repaired and checked by reading, not compiled, not run on hardware
//
#include <hip/hip_runtime.h>
#include <hip/hip_bf16.h>

// ---------- CDNA5 WMMA fragment types ----------
typedef __bf16 bf16_t;
typedef bf16_t v16bf __attribute__((ext_vector_type(16)));
typedef bf16_t v8bf  __attribute__((ext_vector_type(8)));
typedef float  v8f   __attribute__((ext_vector_type(8)));

#define HIDDEN 1024
#define NHEADS 16
#define HEAD   64
#define BATCH  2
#define SEQ    2048
#define ROWS   (BATCH * SEQ)   // 4096
#define QTILES (SEQ / 16)      // 128
#define KCHUNKS (SEQ / 32)     // 64

static __device__ __forceinline__ v8f wmma_bf16(v16bf a, v16bf b, v8f c) {
    return __builtin_amdgcn_wmma_f32_16x16x32_bf16(false, a, false, b, (short)0, c,
                                                   false, false);
}

// Build a 16-wide bf16 A/P fragment from two contiguous 8-half chunks.
static __device__ __forceinline__ v16bf combine8(v8bf lo, v8bf hi) {
    return __builtin_shufflevector(lo, hi, 0, 1, 2, 3, 4, 5, 6, 7,
                                           8, 9, 10, 11, 12, 13, 14, 15);
}

// ---------- Kernel 1: fp32 -> bf16 convert of hidden states ----------
__global__ __launch_bounds__(256) void convert_x_kernel(const float* __restrict__ X,
                                                        bf16_t* __restrict__ Xbf) {
    int i = blockIdx.x * 256 + threadIdx.x;   // ROWS*HIDDEN threads
    Xbf[i] = (bf16_t)X[i];
}

// ---------- Kernel 2: Wt[n][k] = (bf16)W[k][n] for Wq/Wk/Wv ----------
__global__ __launch_bounds__(256) void transpose_w_kernel(const float* __restrict__ Wq,
                                                          const float* __restrict__ Wk,
                                                          const float* __restrict__ Wv,
                                                          bf16_t* __restrict__ WtAll) {
    const float* W = (blockIdx.z == 0) ? Wq : (blockIdx.z == 1) ? Wk : Wv;
    bf16_t* Wt = WtAll + (size_t)blockIdx.z * HIDDEN * HIDDEN;
    __shared__ float tile[16][17];
    int n = blockIdx.x * 16 + threadIdx.x;
    int k = blockIdx.y * 16 + threadIdx.y;
    tile[threadIdx.y][threadIdx.x] = W[(size_t)k * HIDDEN + n];
    __syncthreads();
    int n2 = blockIdx.x * 16 + threadIdx.y;
    int k2 = blockIdx.y * 16 + threadIdx.x;
    Wt[(size_t)n2 * HIDDEN + k2] = (bf16_t)tile[threadIdx.x][threadIdx.y];
}

// ---------- Kernel 3: per-query-tile active key-chunk bitmask ----------
// One wave per query tile (16 rows). Chunk c (32 keys) is active iff any mask
// element in the 16x32 block is not -1e4. 128 waves total.
__global__ __launch_bounds__(256) void scan_mask_kernel(const float* __restrict__ smask,
                                                        unsigned long long* __restrict__ active) {
    int wave = (blockIdx.x * 256 + threadIdx.x) >> 5;   // 0..127 == qtile
    int lane = threadIdx.x & 31;
    int q0 = wave * 16;
    unsigned long long bits = 0;
    for (int c = 0; c < KCHUNKS; ++c) {
        int col = c * 32 + lane;
        int any = 0;
        for (int r = 0; r < 16; ++r) {
            float v = smask[(size_t)(q0 + r) * SEQ + col];
            if (v > -5000.0f) any = 1;
        }
        if (__any(any)) bits |= (1ull << c);
    }
    if (lane == 0) active[wave] = bits;
}

// ---------- Kernel 4: QKV GEMM  [4096x1024] x [1024x1024] via WMMA bf16 ----------
// grid (16, 16, 3), block 256 (8 waves). Wave w: 32 rows at m0 = by*256 + w*32,
// 64 cols at n0 = bx*64 -> 8 accumulators, 8 WMMAs per 32-wide k-step.
// mat 0/1 -> Q/K stored bf16 [b,h,l,d]; mat 2 -> V stored transposed bf16 [b,h,d,l].
__global__ __launch_bounds__(256) void qkv_gemm_kernel(const bf16_t* __restrict__ Xbf,
                                                       const bf16_t* __restrict__ WtAll,
                                                       const float* __restrict__ bq,
                                                       const float* __restrict__ bk,
                                                       const float* __restrict__ bv,
                                                       bf16_t* __restrict__ Qbf,
                                                       bf16_t* __restrict__ Kbf,
                                                       bf16_t* __restrict__ Vt) {
    const int mat = blockIdx.z;
    const bf16_t* Wt = WtAll + (size_t)mat * HIDDEN * HIDDEN;
    const float* bias = (mat == 0) ? bq : (mat == 1) ? bk : bv;

    const int lane = threadIdx.x & 31;
    const int wave = threadIdx.x >> 5;
    const int ln = lane & 15;        // row (A) / col (B,D) within tile
    const int hi = lane >> 4;        // half-wave selector

    const int m0 = blockIdx.y * 256 + wave * 32;
    const int n0 = blockIdx.x * 64;

    v8f acc[2][4];
    for (int u = 0; u < 2; ++u)
        for (int t = 0; t < 4; ++t)
            for (int r = 0; r < 8; ++r) acc[u][t][r] = 0.0f;

    const bf16_t* xrow0 = Xbf + (size_t)(m0 + ln) * HIDDEN + 16 * hi;
    const bf16_t* xrow1 = Xbf + (size_t)(m0 + 16 + ln) * HIDDEN + 16 * hi;
    for (int k0 = 0; k0 < HIDDEN; k0 += 32) {
        // A fragments: K = {k0+16hi..+15}
        v16bf a0 = combine8(*(const v8bf*)(xrow0 + k0), *(const v8bf*)(xrow0 + k0 + 8));
        v16bf a1 = combine8(*(const v8bf*)(xrow1 + k0), *(const v8bf*)(xrow1 + k0 + 8));
        for (int t = 0; t < 4; ++t) {
            int n = n0 + t * 16 + ln;
            // B fragment: column n of W = row n of Wt, 16 contiguous K halves
            v16bf bfrag = *(const v16bf*)(Wt + (size_t)n * HIDDEN + k0 + 16 * hi);
            acc[0][t] = wmma_bf16(a0, bfrag, acc[0][t]);
            acc[1][t] = wmma_bf16(a1, bfrag, acc[1][t]);
        }
    }

    // Epilogue: D element r -> (row m0+u*16+r+8hi, col n0+t*16+ln)
    for (int u = 0; u < 2; ++u) {
        for (int t = 0; t < 4; ++t) {
            for (int r = 0; r < 8; ++r) {
                int gm = m0 + u * 16 + r + 8 * hi;
                int gn = n0 + t * 16 + ln;
                float val = acc[u][t][r] + bias[gn];
                int b = gm >> 11, l = gm & (SEQ - 1);
                int h = gn >> 6, d = gn & (HEAD - 1);
                int bh = b * NHEADS + h;
                if (mat == 2) {
                    Vt[(((size_t)bh * HEAD + d) * SEQ) + l] = (bf16_t)val;
                } else {
                    bf16_t* dst = (mat == 1) ? Kbf : Qbf;
                    dst[(((size_t)bh * SEQ + l) * HEAD) + d] = (bf16_t)val;
                }
            }
        }
    }
}

// ---------- Kernel 5: flash-style sparse-masked attention ----------
// grid (32, 32), block 128 (4 waves). Wave handles query tile qt = bx*4 + w of (b,h)=by.
// Fully-masked 16x32 key chunks are skipped via a wave-uniform scalar branch
// (exact: clipped masked scores give exp(-1e4 - M) == 0 in fp32).
__global__ __launch_bounds__(128) void attn_kernel(const bf16_t* __restrict__ Qbf,
                                                   const bf16_t* __restrict__ Kbf,
                                                   const bf16_t* __restrict__ Vt,
                                                   const float* __restrict__ smask,
                                                   const float* __restrict__ amask,
                                                   const unsigned long long* __restrict__ active,
                                                   float* __restrict__ out) {
    __shared__ bf16_t Pbuf[4][16 * 32];   // per-wave 16x32 bf16 probs tile

    const int lane = threadIdx.x & 31;
    const int wave = threadIdx.x >> 5;
    const int ln = lane & 15;
    const int hi = lane >> 4;

    const int qt = blockIdx.x * 4 + wave;
    const int q0 = qt * 16;
    const int bh = blockIdx.y;
    const int b = bh >> 4, h = bh & 15;

    const bf16_t* Qb = Qbf + (size_t)bh * SEQ * HEAD;
    const bf16_t* Kb = Kbf + (size_t)bh * SEQ * HEAD;
    const bf16_t* Vb = Vt + (size_t)bh * HEAD * SEQ;
    bf16_t* Pw = &Pbuf[wave][0];

    const unsigned long long mbits = active[qt];

    // Q A-fragments for the two 32-wide d-chunks
    v16bf aQ[2];
    for (int c = 0; c < 2; ++c) {
        const bf16_t* p = Qb + (size_t)(q0 + ln) * HEAD + c * 32 + 16 * hi;
        aQ[c] = combine8(*(const v8bf*)p, *(const v8bf*)(p + 8));
    }

    v8f acc[4];
    float rmax[8], rsum[8], alpha[8];
    for (int t = 0; t < 4; ++t)
        for (int r = 0; r < 8; ++r) acc[t][r] = 0.0f;
    for (int r = 0; r < 8; ++r) { rmax[r] = -3.0e38f; rsum[r] = 0.0f; }

    for (int kc = 0; kc < SEQ; kc += 32) {
        // Wave-uniform sparse skip: force condition into an SGPR.
        unsigned act = (unsigned)((mbits >> (kc >> 5)) & 1ull);
        act = __builtin_amdgcn_readfirstlane(act);
        if (!act) continue;

        // ---- scores: two 16x16 tiles, S = Q K^T ----
        v8f S[2];
        for (int t = 0; t < 2; ++t) {
            int col = kc + t * 16 + ln;          // key index for this lane
            v8f s;
            for (int r = 0; r < 8; ++r) s[r] = 0.0f;
            const bf16_t* kp = Kb + (size_t)col * HEAD + 16 * hi;
            v16bf b0 = *(const v16bf*)(kp);       // d chunk 0..31
            v16bf b1 = *(const v16bf*)(kp + 32);  // d chunk 32..63
            s = wmma_bf16(aQ[0], b0, s);
            s = wmma_bf16(aQ[1], b1, s);
            float amv = amask[b * SEQ + col];
            for (int r = 0; r < 8; ++r) {
                int q = q0 + r + 8 * hi;
                float sv = s[r] * 0.125f + smask[(size_t)q * SEQ + col];
                sv = fminf(fmaxf(sv, -10000.0f), 10000.0f) + amv;
                s[r] = sv;
            }
            S[t] = s;
        }
        // ---- online softmax over this 32-key chunk ----
        for (int r = 0; r < 8; ++r) {
            float mx = fmaxf(S[0][r], S[1][r]);
            for (int d = 1; d < 16; d <<= 1) mx = fmaxf(mx, __shfl_xor(mx, d, 32));
            float Mn = fmaxf(rmax[r], mx);
            alpha[r] = __expf(rmax[r] - Mn);
            rmax[r] = Mn;
            float p0 = __expf(S[0][r] - Mn);
            float p1 = __expf(S[1][r] - Mn);
            S[0][r] = p0;
            S[1][r] = p1;
            float ps = p0 + p1;
            for (int d = 1; d < 16; d <<= 1) ps += __shfl_xor(ps, d, 32);
            rsum[r] = alpha[r] * rsum[r] + ps;
        }
        // ---- D-layout -> A-layout via LDS (bf16 probs) ----
        for (int t = 0; t < 2; ++t)
            for (int r = 0; r < 8; ++r)
                Pw[(r + 8 * hi) * 32 + t * 16 + ln] = (bf16_t)S[t][r];
        // rescale running context while LDS writes drain
        for (int t = 0; t < 4; ++t)
            for (int r = 0; r < 8; ++r) acc[t][r] *= alpha[r];
        asm volatile("s_wait_dscnt 0" ::: "memory");
        const bf16_t* pp = Pw + ln * 32 + 16 * hi;
        v16bf aP = combine8(*(const v8bf*)pp, *(const v8bf*)(pp + 8));
        // ---- ctx += P x V ----
        for (int t = 0; t < 4; ++t) {
            int d = t * 16 + ln;
            v16bf bv = *(const v16bf*)(Vb + (size_t)d * SEQ + kc + 16 * hi);
            acc[t] = wmma_bf16(aP, bv, acc[t]);
        }
    }

    // ---- epilogue: normalize, merge heads -> out[b, q, h*64 + d] ----
    for (int t = 0; t < 4; ++t) {
        for (int r = 0; r < 8; ++r) {
            int q = q0 + r + 8 * hi;
            float o = acc[t][r] / rsum[r];
            out[((size_t)(b * SEQ + q)) * HIDDEN + h * HEAD + t * 16 + ln] = o;
        }
    }
}

// ---------- Host launcher ----------
extern "C" void kernel_launch(void* const* d_in, const int* in_sizes, int n_in,
                              void* d_out, int out_size, void* d_ws, size_t ws_size,
                              hipStream_t stream) {
    const float* X     = (const float*)d_in[0];
    const float* amask = (const float*)d_in[1];
    const float* Wq    = (const float*)d_in[2];
    const float* bq    = (const float*)d_in[3];
    const float* Wk    = (const float*)d_in[4];
    const float* bk    = (const float*)d_in[5];
    const float* Wv    = (const float*)d_in[6];
    const float* bv    = (const float*)d_in[7];
    const float* smask = (const float*)d_in[8];

    char* ws = (char*)d_ws;
    const size_t SZ_X  = (size_t)ROWS * HIDDEN * sizeof(bf16_t);       // 8 MB
    const size_t SZ_W  = (size_t)3 * HIDDEN * HIDDEN * sizeof(bf16_t); // 6 MB
    const size_t SZ_QK = (size_t)BATCH * NHEADS * SEQ * HEAD * sizeof(bf16_t); // 8 MB
    bf16_t* Xbf = (bf16_t*)(ws);
    bf16_t* Wt  = (bf16_t*)(ws + SZ_X);
    bf16_t* Qbf = (bf16_t*)(ws + SZ_X + SZ_W);
    bf16_t* Kbf = (bf16_t*)(ws + SZ_X + SZ_W + SZ_QK);
    bf16_t* Vt  = (bf16_t*)(ws + SZ_X + SZ_W + 2 * SZ_QK);
    unsigned long long* active =
        (unsigned long long*)(ws + SZ_X + SZ_W + 3 * SZ_QK);           // 1 KB

    convert_x_kernel<<<(ROWS * HIDDEN) / 256, 256, 0, stream>>>(X, Xbf);
    transpose_w_kernel<<<dim3(HIDDEN / 16, HIDDEN / 16, 3), dim3(16, 16), 0, stream>>>(
        Wq, Wk, Wv, Wt);
    scan_mask_kernel<<<QTILES / 8, 256, 0, stream>>>(smask, active);
    qkv_gemm_kernel<<<dim3(HIDDEN / 64, ROWS / 256, 3), 256, 0, stream>>>(
        Xbf, Wt, bq, bk, bv, Qbf, Kbf, Vt);
    attn_kernel<<<dim3(QTILES / 4, BATCH * NHEADS), 128, 0, stream>>>(
        Qbf, Kbf, Vt, smask, amask, active, (float*)d_out);
}
